// SelfAttention_78829829751108
// MI455X (gfx1250) — matrix operand order repaired and checked by reading, not compiled, not run on hardware
//
#include <hip/hip_runtime.h>
#include <stdint.h>

#define B_   2
#define T_   2048
#define EMB_ 1024
#define H_   16
#define S_   64
#define M_   (B_*T_)   // 4096 rows

typedef __attribute__((ext_vector_type(16))) __bf16   v16bf;
typedef __attribute__((ext_vector_type(8)))  float    v8f;
typedef __attribute__((ext_vector_type(4)))  uint32_t v4u;
typedef unsigned short u16;

// ---------- helpers ----------

__device__ __forceinline__ u16 f2bf(float f) {
  union { float f; uint32_t u; } c; c.f = f;
  uint32_t u = c.u;
  u += 0x7FFFu + ((u >> 16) & 1u);   // round-to-nearest-even
  return (u16)(u >> 16);
}

// A-fragment (16x32 bf16): lane = row (lane%16); elements 0..7 at K = g*8..g*8+7,
// elements 8..15 at K = g*8+16..g*8+23  (ISA 7.12.2, 16-bit A 16x32)
__device__ __forceinline__ v16bf load_fragA(const u16* p) {
  union { v4u q[2]; v16bf v; } r;
  r.q[0] = *(const v4u*)(p);
  r.q[1] = *(const v4u*)(p + 16);
  return r.v;
}
// B-fragment (32x16 bf16): lane = col (lane%16); elements 0..7 at K = g*8..g*8+7, 8..15 at K = g*8+16..g*8+23
__device__ __forceinline__ v16bf load_fragB(const u16* p) {
  union { v4u q[2]; v16bf v; } r;
  r.q[0] = *(const v4u*)(p);
  r.q[1] = *(const v4u*)(p + 16);
  return r.v;
}
// B-fragment read from LDS via an explicit addrspace(3) pointer forged from the
// LDS byte offset -> guaranteed ds_load_b128 (DS ops have no scope field).
// volatile: the tile is written by async-copy inline asm the optimizer can't see.
__device__ __forceinline__ v16bf load_fragB_lds_off(uint32_t byte_off) {
  typedef __attribute__((address_space(3))) const volatile v4u* lds_p;
  lds_p p = (lds_p)byte_off;
  union { v4u q[2]; v16bf v; } r;
  r.q[0] = p[0];
  r.q[1] = p[2];
  return r.v;
}

#define WMMA_BF16(a, b, c) \
  __builtin_amdgcn_wmma_f32_16x16x32_bf16(false, (a), false, (b), (short)0, (c), false, false)

#define WAIT_ASYNC0() asm volatile("s_wait_asynccnt 0x0" ::: "memory")
#define WAIT_ASYNC1() asm volatile("s_wait_asynccnt 0x1" ::: "memory")

// Async copy 16B global -> LDS (CDNA5 GLOBAL_LOAD_ASYNC_TO_LDS_B128, ASYNCcnt)
__device__ __forceinline__ void async_g2l_b128(uint32_t lds_byte_off, const u16* gaddr) {
  asm volatile("global_load_async_to_lds_b128 %0, %1, off"
               :: "v"(lds_byte_off), "v"(gaddr) : "memory");
}

// ---------- kernel 1: fp32 -> bf16 ----------
__global__ void __launch_bounds__(256)
cvt_bf16(const float* __restrict__ src, u16* __restrict__ dst, int n) {
  int i = blockIdx.x * 256 + threadIdx.x;
  if (i < n) dst[i] = f2bf(src[i]);
}

// ---------- kernel 2: W[k][n] -> Wt[n][k] bf16 (tiled) ----------
__global__ void __launch_bounds__(256)
transpose_cvt(const float* __restrict__ W, u16* __restrict__ Wt) {
  __shared__ u16 tile[32][33];
  int x  = blockIdx.x * 32 + threadIdx.x;   // col n of W
  int y0 = blockIdx.y * 32;                 // row k base
  for (int j = threadIdx.y; j < 32; j += 8)
    tile[j][threadIdx.x] = f2bf(W[(size_t)(y0 + j) * EMB_ + x]);
  __syncthreads();
  int n0 = blockIdx.x * 32;
  for (int j = threadIdx.y; j < 32; j += 8)
    Wt[(size_t)(n0 + j) * EMB_ + y0 + threadIdx.x] = tile[threadIdx.x][j];
}

// ---------- kernel 3: bf16 WMMA GEMM: C[m,n] = sum_k A[m,k]*Wt[n,k] ----------
// B panel (64 cols, shared by all 8 waves) triple-buffered in LDS via async
// global->LDS copies, prefetch distance 2, steady-state loop is branch-free
// (tail iterations peeled). A fragments register double-buffered.
// mode 0: bf16 row-major out (Q,K); 1: bf16 Vt[b][h][s][t]; 2: fp32 + bias
#define BROW_ 40                    // padded LDS row stride (elems): banks 4 apart
#define BBUF_ (64 * BROW_ * 2)      // 5120 bytes per buffer
#define NBUF_ 3
__global__ void __launch_bounds__(256)
gemm_bf16_wmma(const u16* __restrict__ A, const u16* __restrict__ Bt,
               int mode, u16* __restrict__ outb, float* __restrict__ outf,
               const float* __restrict__ bias) {
  __shared__ u16 smem[NBUF_][64 * BROW_];
  const int K = EMB_, N = EMB_;
  int tid  = threadIdx.x;
  int wid  = tid >> 5;
  int lane = tid & 31;
  int l16  = lane & 15;
  int g    = lane >> 4;
  int m_blk = blockIdx.x / (N / 64);
  int n_blk = blockIdx.x % (N / 64);
  int tm = m_blk * 128 + wid * 16;   // this wave: 16 rows x 64 cols
  int tn = n_blk * 64;

  // low 32 bits of a generic pointer into LDS == LDS byte offset (aperture rule)
  uint32_t ldsbase = (uint32_t)(uintptr_t)(&smem[0][0]);
  // staging assignment: each thread copies one 16B chunk of the 64x32 B tile
  int srow = tid >> 2;               // 0..63 (col n within panel)
  int sko  = (tid & 3) * 8;          // 0,8,16,24 (k offset)
  const u16* sgp0 = Bt + (size_t)(tn + srow) * K + sko;
  uint32_t srel = ldsbase + (uint32_t)(srow * (BROW_ * 2) + sko * 2);
  // per-wave LDS read base (byte offset of this wave's first fragment row)
  uint32_t rbase = ldsbase + (uint32_t)(l16 * (BROW_ * 2) + g * 16);

  v8f zf = {0.f,0.f,0.f,0.f,0.f,0.f,0.f,0.f};
  v8f acc[4];
  #pragma unroll
  for (int c = 0; c < 4; c++) acc[c] = zf;

  const u16* Arow = A + (size_t)(tm + l16) * K;

  auto do_tile = [&](int bufi, v16bf afv) {
    uint32_t boff = rbase + (uint32_t)bufi * BBUF_;
    #pragma unroll
    for (int c = 0; c < 4; c++) {
      v16bf bf = load_fragB_lds_off(boff + (uint32_t)(c * 16 * BROW_ * 2));
      acc[c] = WMMA_BF16(afv, bf, acc[c]);
    }
  };

  // prologue: stage k-tiles 0 and 1, prefetch first A fragment
  async_g2l_b128(srel + 0 * BBUF_, sgp0);
  async_g2l_b128(srel + 1 * BBUF_, sgp0 + 32);
  v16bf af = load_fragA(Arow + g * 8);
  WAIT_ASYNC1();      // tile 0 landed; tile 1 still in flight
  __syncthreads();

  // steady state: tiles 0..29 (stage t+2, branch-free)
  int buf = 0;
  for (int kk = 0; kk + 64 < K; kk += 32) {
    int nb = buf + 2; if (nb >= NBUF_) nb -= NBUF_;
    async_g2l_b128(srel + (uint32_t)nb * BBUF_, sgp0 + kk + 64);
    v16bf afn = load_fragA(Arow + kk + 32 + g * 8);   // register prefetch
    do_tile(buf, af);
    WAIT_ASYNC1();    // tile kk+32 has landed; tile kk+64 still in flight
    __syncthreads();
    af = afn;
    buf += 1; if (buf >= NBUF_) buf -= NBUF_;
  }
  // tile 30 (kk = K-64): nothing left to stage
  {
    v16bf afn = load_fragA(Arow + (K - 32) + g * 8);
    do_tile(buf, af);
    WAIT_ASYNC0();    // final tile landed
    __syncthreads();
    af = afn;
    buf += 1; if (buf >= NBUF_) buf -= NBUF_;
  }
  // tile 31 (kk = K-32)
  do_tile(buf, af);

  if (mode == 0) {
    #pragma unroll
    for (int c = 0; c < 4; c++) {
      int n = tn + c * 16 + l16;
      #pragma unroll
      for (int r = 0; r < 8; r++)
        outb[(size_t)(tm + g * 8 + r) * N + n] = f2bf(acc[c][r]);
    }
  } else if (mode == 1) {
    int m0 = tm + g * 8;
    int batch = m0 / T_;
    int t0 = m0 % T_;
    #pragma unroll
    for (int c = 0; c < 4; c++) {
      int n = tn + c * 16 + l16;
      int h = n >> 6, s = n & 63;
      union { u16 h8[8]; v4u q; } pk;
      #pragma unroll
      for (int r = 0; r < 8; r++) pk.h8[r] = f2bf(acc[c][r]);
      *(v4u*)(outb + (size_t)((batch * H_ + h) * S_ + s) * T_ + t0) = pk.q;
    }
  } else {
    #pragma unroll
    for (int c = 0; c < 4; c++) {
      int n = tn + c * 16 + l16;
      float bv = bias[n];
      #pragma unroll
      for (int r = 0; r < 8; r++)
        outf[(size_t)(tm + g * 8 + r) * N + n] = acc[c][r] + bv;
    }
  }
}

// ---------- kernel 4: flash attention (one wave = 16 queries of one (b,h)) ----------
__global__ void __launch_bounds__(256)
attn_wmma(const u16* __restrict__ Qb, const u16* __restrict__ Kb,
          const u16* __restrict__ Vt, u16* __restrict__ Ob) {
  const float scale = 0.125f;   // 1/sqrt(64)
  int wid  = threadIdx.x >> 5;
  int lane = threadIdx.x & 31;
  int l16  = lane & 15;
  int g    = lane >> 4;
  int gwid = blockIdx.x * 8 + wid;
  int qt = gwid & (T_ / 16 - 1);
  int bh = gwid >> 7;
  int b  = bh >> 4;
  int h  = bh & 15;
  int qbase = qt * 16;

  // Q as B-operand fragments for S^T = K * Q^T (kept in regs for whole key loop)
  const u16* Qrow = Qb + (size_t)(b * T_ + qbase + l16) * EMB_ + h * S_;
  v16bf qf[2];
  #pragma unroll
  for (int ks = 0; ks < 2; ks++) qf[ks] = load_fragB(Qrow + ks * 32 + g * 8);

  v8f zf = {0.f,0.f,0.f,0.f,0.f,0.f,0.f,0.f};
  v8f o[4];
  #pragma unroll
  for (int c = 0; c < 4; c++) o[c] = zf;
  float mrow = -__builtin_inff();   // per-q running max (q = qbase + lane%16)
  float lrow = 0.f;                 // per-q running denom

  int nch = (qbase >> 5) + 1;       // 32-key chunks, causal bound
  for (int ch = 0; ch < nch; ch++) {
    int j0 = ch * 32;
    // S^T[key, q]: D layout == A-fragment layout for the PV matmul
    v8f st[2]; st[0] = zf; st[1] = zf;
    #pragma unroll
    for (int tile = 0; tile < 2; tile++) {
      const u16* Krow = Kb + (size_t)(b * T_ + j0 + tile * 16 + l16) * EMB_ + h * S_;
      #pragma unroll
      for (int ks = 0; ks < 2; ks++) {
        v16bf kf = load_fragA(Krow + ks * 32 + g * 8);
        st[tile] = WMMA_BF16(kf, qf[ks], st[tile]);
      }
    }
    // issue V-fragment loads now: in-order load returns mean the PV wmma's
    // wait covers only these, and the softmax VALU below hides their latency
    v16bf vf[4];
    #pragma unroll
    for (int c = 0; c < 4; c++) {
      const u16* Vrow = Vt + (size_t)((b * H_ + h) * S_ + (c * 16 + l16)) * T_ + j0;
      vf[c] = load_fragB(Vrow + g * 8);
    }
    // scale + causal mask + chunk max (per q: 16 local vals + xor-16 partner)
    int q = qbase + l16;
    float cm = -__builtin_inff();
    #pragma unroll
    for (int tile = 0; tile < 2; tile++) {
      #pragma unroll
      for (int r = 0; r < 8; r++) {
        int key = j0 + tile * 16 + g * 8 + r;
        float v = st[tile][r] * scale;
        v = (key <= q) ? v : -__builtin_inff();
        st[tile][r] = v;
        cm = fmaxf(cm, v);
      }
    }
    cm = fmaxf(cm, __shfl_xor(cm, 16, 32));
    float mnew  = fmaxf(mrow, cm);
    float alpha = __expf(mrow - mnew);
    mrow = mnew;

    // P = exp(S - m); st[0] regs -> A elements 0..7, st[1] -> 8..15
    float rs = 0.f;
    union { u16 hh[16]; v16bf v; } pa;
    #pragma unroll
    for (int i = 0; i < 8; i++) {
      float e0 = __expf(st[0][i] - mnew);
      float e1 = __expf(st[1][i] - mnew);
      rs += e0 + e1;
      pa.hh[i]     = f2bf(e0);
      pa.hh[i + 8] = f2bf(e1);
    }
    rs += __shfl_xor(rs, 16, 32);
    lrow = lrow * alpha + rs;

    // rescale O rows: row of O reg r is q = qbase + g*8 + r
    #pragma unroll
    for (int r = 0; r < 8; r++) {
      float af = __shfl(alpha, g * 8 + r, 32);
      #pragma unroll
      for (int c = 0; c < 4; c++) o[c][r] *= af;
    }
    // O += P * V
    #pragma unroll
    for (int c = 0; c < 4; c++) o[c] = WMMA_BF16(pa.v, vf[c], o[c]);
  }

  // normalize by l and store bf16 row-major (feeds final projection GEMM)
  #pragma unroll
  for (int r = 0; r < 8; r++) {
    float lr   = __shfl(lrow, g * 8 + r, 32);
    float linv = 1.0f / lr;
    size_t m = (size_t)(b * T_ + qbase + g * 8 + r);
    #pragma unroll
    for (int c = 0; c < 4; c++)
      Ob[m * EMB_ + h * S_ + c * 16 + l16] = f2bf(o[c][r] * linv);
  }
}

// ---------- host ----------
extern "C" void kernel_launch(void* const* d_in, const int* in_sizes, int n_in,
                              void* d_out, int out_size, void* d_ws, size_t ws_size,
                              hipStream_t stream) {
  (void)in_sizes; (void)n_in; (void)out_size; (void)ws_size;
  const float* x  = (const float*)d_in[0];
  const float* Wq = (const float*)d_in[1];
  const float* Wk = (const float*)d_in[2];
  const float* Wv = (const float*)d_in[3];
  const float* Wu = (const float*)d_in[4];
  const float* bu = (const float*)d_in[5];

  const size_t NX = (size_t)M_ * EMB_;    // 4,194,304
  const size_t NW = (size_t)EMB_ * EMB_;  // 1,048,576
  u16* xb   = (u16*)d_ws;
  u16* Wq_t = xb   + NX;
  u16* Wk_t = Wq_t + NW;
  u16* Wv_t = Wk_t + NW;
  u16* Wu_t = Wv_t + NW;
  u16* Qb   = Wu_t + NW;
  u16* Kb   = Qb   + NX;
  u16* Vt   = Kb   + NX;
  u16* Ob   = Vt   + NX;

  cvt_bf16<<<(int)(NX / 256), 256, 0, stream>>>(x, xb, (int)NX);
  dim3 tb(32, 8), tg(32, 32);
  transpose_cvt<<<tg, tb, 0, stream>>>(Wq, Wq_t);
  transpose_cvt<<<tg, tb, 0, stream>>>(Wk, Wk_t);
  transpose_cvt<<<tg, tb, 0, stream>>>(Wv, Wv_t);
  transpose_cvt<<<tg, tb, 0, stream>>>(Wu, Wu_t);

  const int GB = (M_ / 128) * (EMB_ / 64);  // 512 blocks, 8 waves each
  gemm_bf16_wmma<<<GB, 256, 0, stream>>>(xb, Wq_t, 0, Qb, nullptr, nullptr);
  gemm_bf16_wmma<<<GB, 256, 0, stream>>>(xb, Wk_t, 0, Kb, nullptr, nullptr);
  gemm_bf16_wmma<<<GB, 256, 0, stream>>>(xb, Wv_t, 1, Vt, nullptr, nullptr);

  attn_wmma<<<(B_ * H_ * (T_ / 16)) / 8, 256, 0, stream>>>(Qb, Kb, Vt, Ob);

  gemm_bf16_wmma<<<GB, 256, 0, stream>>>(Ob, Wu_t, 2, nullptr, (float*)d_out, bu);
}
